// MLGANN_77584289235308
// MI455X (gfx1250) — matrix operand order repaired and checked by reading, not compiled, not run on hardware
//
#include <hip/hip_runtime.h>

typedef __attribute__((ext_vector_type(16))) _Float16 v16h;
typedef __attribute__((ext_vector_type(8)))  float    v8f;

#define HDIM 128
#define EPS  1e-5f
#define PANEL_ELEMS (8 * 4 * 32 * 16)   // colblocks * ksteps * lanes * 16 halves

// ---------------------------------------------------------------------------
// degree: deg[dst] += 1 for every edge (self-loop +1 folded into dinv kernel)
// ---------------------------------------------------------------------------
__global__ __launch_bounds__(256) void deg_kernel(const int* __restrict__ dst,
                                                  float* __restrict__ deg, int E) {
  int i = blockIdx.x * 256 + threadIdx.x;
  if (i < E) unsafeAtomicAdd(&deg[dst[i]], 1.0f);
}

__global__ __launch_bounds__(256) void dinv_kernel(float* __restrict__ deg, int N) {
  int i = blockIdx.x * 256 + threadIdx.x;
  if (i < N) deg[i] = rsqrtf(deg[i] + 1.0f);   // deg includes self-loop, always > 0
}

// ---------------------------------------------------------------------------
// Pre-convert a 128x128 f32 weight (row-major [in,out]) into f16 WMMA B-fragment
// order: panel[((cb*4 + s)*32 + lane)*16 + j] = B[(32s + (lane>>4)*16 + j)*128 + cb*16 + (lane&15)]
// ---------------------------------------------------------------------------
__global__ __launch_bounds__(256) void prep_bpanel(const float* __restrict__ B,
                                                   _Float16* __restrict__ panel) {
  int idx = blockIdx.x * 256 + threadIdx.x;
  if (idx >= PANEL_ELEMS) return;
  int j  = idx & 15;
  int l  = (idx >> 4) & 31;
  int s  = (idx >> 9) & 3;
  int cb = idx >> 11;
  int k  = 32 * s + (l >> 4) * 16 + j;
  int n  = cb * 16 + (l & 15);
  panel[idx] = (_Float16)B[(size_t)k * HDIM + n];
}

// ---------------------------------------------------------------------------
// WMMA GEMM: out[M,128] = A[M,128] @ B[128,128] (+bias), B pre-packed as panel.
// Block owns a 32-row strip (two 16-row tiles so each B fragment feeds 2 WMMAs);
// wave w owns columns 16w..16w+15. A strip staged through LDS in fragment order.
// M must be a multiple of 16 (tail tile handled by uniform guard).
// ---------------------------------------------------------------------------
__global__ __launch_bounds__(256) void gemm128_wmma(const float* __restrict__ A,
                                                    const _Float16* __restrict__ Bp,
                                                    const float* __restrict__ bias,
                                                    float* __restrict__ out, int M) {
  __shared__ __align__(32) _Float16 Al[2][4 * 32 * 16];   // [tile][kstep][lane][16]
  const int row0 = blockIdx.x << 5;
  if (row0 >= M) return;
  const bool has2 = (row0 + 16) < M;     // block-uniform
  const int t = threadIdx.x;

  // cooperative A-strip load + f32->f16 convert into fragment layout
  {
    int jh  = t & 1;           // which 8-half group of the 16-half fragment
    int l   = (t >> 1) & 31;   // fragment lane
    int s   = t >> 6;          // kstep
    int m   = l & 15;
    int kh2 = l >> 4;
    // k = 32s + kh2*16 + j
    size_t coff = 32 * s + 16 * kh2 + jh * 8;
    _Float16* d0 = &Al[0][((s * 32 + l) << 4) + jh * 8];
    const float* p0 = A + (size_t)(row0 + m) * HDIM + coff;
    float4 f0 = *(const float4*)p0;
    float4 f1 = *(const float4*)(p0 + 4);
    d0[0] = (_Float16)f0.x; d0[1] = (_Float16)f0.y; d0[2] = (_Float16)f0.z; d0[3] = (_Float16)f0.w;
    d0[4] = (_Float16)f1.x; d0[5] = (_Float16)f1.y; d0[6] = (_Float16)f1.z; d0[7] = (_Float16)f1.w;
    if (has2) {
      _Float16* d1 = &Al[1][((s * 32 + l) << 4) + jh * 8];
      const float* p1 = A + (size_t)(row0 + 16 + m) * HDIM + coff;
      float4 g0 = *(const float4*)p1;
      float4 g1 = *(const float4*)(p1 + 4);
      d1[0] = (_Float16)g0.x; d1[1] = (_Float16)g0.y; d1[2] = (_Float16)g0.z; d1[3] = (_Float16)g0.w;
      d1[4] = (_Float16)g1.x; d1[5] = (_Float16)g1.y; d1[6] = (_Float16)g1.z; d1[7] = (_Float16)g1.w;
    }
  }
  __syncthreads();

  const int wave = t >> 5;
  const int lane = t & 31;
  const int col0 = wave << 4;
  const v16h* bp = (const v16h*)Bp + (size_t)wave * 4 * 32;   // panel for this colblock

  v8f acc0 = {};
  v8f acc1 = {};
#pragma unroll
  for (int s = 0; s < 4; ++s) {
    v16h bv = bp[s * 32 + lane];
    v16h a0 = *(const v16h*)&Al[0][(s * 32 + lane) << 4];
    acc0 = __builtin_amdgcn_wmma_f32_16x16x32_f16(false, a0, false, bv,
                                                  (short)0, acc0, false, false);
    if (has2) {
      v16h a1 = *(const v16h*)&Al[1][(s * 32 + lane) << 4];
      acc1 = __builtin_amdgcn_wmma_f32_16x16x32_f16(false, a1, false, bv,
                                                    (short)0, acc1, false, false);
    }
  }

  const int m  = lane & 15;
  const int kh = lane >> 4;
  const float bb = bias ? bias[col0 + m] : 0.0f;
#pragma unroll
  for (int r = 0; r < 8; ++r)
    out[(size_t)(row0 + r + 8 * kh) * HDIM + col0 + m] = acc0[r] + bb;
  if (has2) {
#pragma unroll
    for (int r = 0; r < 8; ++r)
      out[(size_t)(row0 + 16 + r + 8 * kh) * HDIM + col0 + m] = acc1[r] + bb;
  }
}

// ---------------------------------------------------------------------------
// agg init with self-loop term: agg = hw * dinv[node]^2   (float4 per thread)
// ---------------------------------------------------------------------------
__global__ __launch_bounds__(256) void selfloop_init(const float* __restrict__ hw,
                                                     const float* __restrict__ dinv,
                                                     float* __restrict__ agg, int quads) {
  int i = blockIdx.x * 256 + threadIdx.x;   // one float4 per thread
  if (i >= quads) return;
  float d = dinv[i >> 5];                   // 32 float4s per node row
  float w = d * d;
  float4 v = ((const float4*)hw)[i];
  v.x *= w; v.y *= w; v.z *= w; v.w *= w;
  ((float4*)agg)[i] = v;
}

// ---------------------------------------------------------------------------
// edge scatter: one wave per edge, float4 per lane, f32 hardware atomics
// ---------------------------------------------------------------------------
__global__ __launch_bounds__(256) void edge_scatter(const float* __restrict__ hw,
                                                    const int* __restrict__ src,
                                                    const int* __restrict__ dst,
                                                    const float* __restrict__ dinv,
                                                    float* __restrict__ agg, int E) {
  int wid  = (blockIdx.x * 256 + threadIdx.x) >> 5;
  int lane = threadIdx.x & 31;
  if (wid >= E) return;
  int s = src[wid], d = dst[wid];
  float w = dinv[s] * dinv[d];
  float4 v = *(const float4*)(hw + (size_t)s * HDIM + lane * 4);
  float* o = agg + (size_t)d * HDIM + lane * 4;
  unsafeAtomicAdd(o + 0, v.x * w);
  unsafeAtomicAdd(o + 1, v.y * w);
  unsafeAtomicAdd(o + 2, v.z * w);
  unsafeAtomicAdd(o + 3, v.w * w);
}

__device__ __forceinline__ float wave_sum(float p) {
#pragma unroll
  for (int off = 1; off < 32; off <<= 1) p += __shfl_xor(p, off, 32);
  return p;
}

// ---------------------------------------------------------------------------
// per-node: new = LN(relu(agg + b)); cur = cur_in + new; emb = cur
// ---------------------------------------------------------------------------
__global__ __launch_bounds__(256) void post_ln(const float* __restrict__ agg,
                                               const float* __restrict__ gb,
                                               const float* __restrict__ lng,
                                               const float* __restrict__ lnb,
                                               const float* __restrict__ cur_in,
                                               float* __restrict__ cur_out,
                                               float* __restrict__ emb_out, int N) {
  int wid  = (blockIdx.x * 256 + threadIdx.x) >> 5;
  int lane = threadIdx.x & 31;
  if (wid >= N) return;
  size_t base = (size_t)wid * HDIM + lane * 4;
  float4 a = *(const float4*)(agg + base);
  float4 b = *(const float4*)(gb + lane * 4);
  float4 x;
  x.x = fmaxf(a.x + b.x, 0.0f); x.y = fmaxf(a.y + b.y, 0.0f);
  x.z = fmaxf(a.z + b.z, 0.0f); x.w = fmaxf(a.w + b.w, 0.0f);

  float mu = wave_sum(x.x + x.y + x.z + x.w) * (1.0f / 128.0f);
  float4 c = {x.x - mu, x.y - mu, x.z - mu, x.w - mu};
  float var = wave_sum(c.x * c.x + c.y * c.y + c.z * c.z + c.w * c.w) * (1.0f / 128.0f);
  float rs = rsqrtf(var + EPS);

  float4 g  = *(const float4*)(lng + lane * 4);
  float4 bl = *(const float4*)(lnb + lane * 4);
  float4 ci = *(const float4*)(cur_in + base);
  float4 o;
  o.x = ci.x + c.x * rs * g.x + bl.x;
  o.y = ci.y + c.y * rs * g.y + bl.y;
  o.z = ci.z + c.z * rs * g.z + bl.z;
  o.w = ci.w + c.w * rs * g.w + bl.w;
  *(float4*)(cur_out + base) = o;
  *(float4*)(emb_out + base) = o;
}

// ---------------------------------------------------------------------------
// per-node 8-head attention over L=3 layers (only l=2 query row needed) + LN.
// ---------------------------------------------------------------------------
__global__ __launch_bounds__(256) void attn_ln(const float* __restrict__ q2,
                                               const float* __restrict__ k,
                                               const float* __restrict__ v,
                                               const float* __restrict__ lng,
                                               const float* __restrict__ lnb,
                                               float* __restrict__ z2, int N) {
  int wid  = (blockIdx.x * 256 + threadIdx.x) >> 5;
  int lane = threadIdx.x & 31;
  if (wid >= N) return;
  size_t base = (size_t)wid * HDIM + lane * 4;
  size_t slab = (size_t)N * HDIM;

  float4 q  = *(const float4*)(q2 + base);
  float4 k0 = *(const float4*)(k + base);
  float4 k1 = *(const float4*)(k + slab + base);
  float4 k2 = *(const float4*)(k + 2 * slab + base);
  float4 v0 = *(const float4*)(v + base);
  float4 v1 = *(const float4*)(v + slab + base);
  float4 v2 = *(const float4*)(v + 2 * slab + base);

  float d0 = q.x * k0.x + q.y * k0.y + q.z * k0.z + q.w * k0.w;
  float d1 = q.x * k1.x + q.y * k1.y + q.z * k1.z + q.w * k1.w;
  float d2 = q.x * k2.x + q.y * k2.y + q.z * k2.z + q.w * k2.w;
  d0 += __shfl_xor(d0, 1, 32); d0 += __shfl_xor(d0, 2, 32);
  d1 += __shfl_xor(d1, 1, 32); d1 += __shfl_xor(d1, 2, 32);
  d2 += __shfl_xor(d2, 1, 32); d2 += __shfl_xor(d2, 2, 32);

  float s0 = d0 * 0.25f, s1 = d1 * 0.25f, s2 = d2 * 0.25f;  // 1/sqrt(16)
  float mx = fmaxf(s0, fmaxf(s1, s2));
  float e0 = __expf(s0 - mx), e1 = __expf(s1 - mx), e2 = __expf(s2 - mx);
  float inv = 1.0f / (e0 + e1 + e2);
  float a0 = e0 * inv, a1 = e1 * inv, a2 = e2 * inv;

  float4 ctx;
  ctx.x = a0 * v0.x + a1 * v1.x + a2 * v2.x;
  ctx.y = a0 * v0.y + a1 * v1.y + a2 * v2.y;
  ctx.z = a0 * v0.z + a1 * v1.z + a2 * v2.z;
  ctx.w = a0 * v0.w + a1 * v1.w + a2 * v2.w;

  float mu = wave_sum(ctx.x + ctx.y + ctx.z + ctx.w) * (1.0f / 128.0f);
  float4 c = {ctx.x - mu, ctx.y - mu, ctx.z - mu, ctx.w - mu};
  float var = wave_sum(c.x * c.x + c.y * c.y + c.z * c.z + c.w * c.w) * (1.0f / 128.0f);
  float rs = rsqrtf(var + EPS);

  float4 g  = *(const float4*)(lng + lane * 4);
  float4 bl = *(const float4*)(lnb + lane * 4);
  float4 y;
  y.x = c.x * rs * g.x + bl.x; y.y = c.y * rs * g.y + bl.y;
  y.z = c.z * rs * g.z + bl.z; y.w = c.w * rs * g.w + bl.w;
  *(float4*)(z2 + base) = y;
}

// ---------------------------------------------------------------------------
// row gather for the 4 id sets, wave per output row
// ---------------------------------------------------------------------------
__global__ __launch_bounds__(256) void gather_rows(const float* __restrict__ zout,
                                                   const int* __restrict__ i0,
                                                   const int* __restrict__ i1,
                                                   const int* __restrict__ i2,
                                                   const int* __restrict__ i3,
                                                   float* __restrict__ out, int P) {
  int wid  = (blockIdx.x * 256 + threadIdx.x) >> 5;
  int lane = threadIdx.x & 31;
  if (wid >= 4 * P) return;
  int set = wid / P;
  int p   = wid - set * P;
  const int* ids = (set == 0) ? i0 : (set == 1) ? i1 : (set == 2) ? i2 : i3;
  int idx = ids[p];
  float4 val = *(const float4*)(zout + (size_t)idx * HDIM + lane * 4);
  *(float4*)(out + (size_t)wid * HDIM + lane * 4) = val;
}

// ---------------------------------------------------------------------------
extern "C" void kernel_launch(void* const* d_in, const int* in_sizes, int n_in,
                              void* d_out, int out_size, void* d_ws, size_t ws_size,
                              hipStream_t stream) {
  const float* x    = (const float*)d_in[0];
  const int*   ei   = (const int*)d_in[1];
  const int*   dpos = (const int*)d_in[2];
  const int*   tpos = (const int*)d_in[3];
  const int*   dneg = (const int*)d_in[4];
  const int*   tneg = (const int*)d_in[5];
  const float* w0   = (const float*)d_in[7];
  const float* b0   = (const float*)d_in[8];
  const float* w1   = (const float*)d_in[9];
  const float* b1   = (const float*)d_in[10];
  const float* w2   = (const float*)d_in[11];
  const float* b2   = (const float*)d_in[12];
  const float* wq   = (const float*)d_in[13];
  const float* wk   = (const float*)d_in[14];
  const float* wv   = (const float*)d_in[15];
  const float* lng  = (const float*)d_in[16];
  const float* lnb  = (const float*)d_in[17];
  const float* ow   = (const float*)d_in[18];
  const float* ob   = (const float*)d_in[19];

  const int N = in_sizes[0] / HDIM;
  const int E = in_sizes[1] / 2;
  const int P = in_sizes[2];
  const int* src = ei;
  const int* dst = ei + E;

  float* ws = (float*)d_ws;
  size_t slab  = (size_t)N * HDIM;
  size_t degsz = ((size_t)N + 63) & ~(size_t)63;   // keep slabs 16B-aligned
  float* deg   = ws;                 // N (becomes dinv)
  float* s_cur = ws + degsz;         // 1 slab
  float* s_emb = s_cur + slab;       // 3 slabs [L,N,128]
  float* s_k   = s_emb + 3 * slab;   // 3 slabs
  float* s_v   = s_k + 3 * slab;     // 3 slabs
  float* s_q   = s_v + 3 * slab;     // 1 slab
  _Float16* panels = (_Float16*)(s_q + slab);  // 7 weight panels, f16 fragment order
  float* s_hw  = s_k;                // alias: dead until k-GEMM
  float* s_agg = s_k + slab;         // alias
  float* s_z   = s_cur;              // alias: cur dead after layer loop
  float* s_out = s_k;                // alias: k dead after attention

  _Float16* p_w0 = panels + 0 * (size_t)PANEL_ELEMS;
  _Float16* p_w1 = panels + 1 * (size_t)PANEL_ELEMS;
  _Float16* p_w2 = panels + 2 * (size_t)PANEL_ELEMS;
  _Float16* p_wk = panels + 3 * (size_t)PANEL_ELEMS;
  _Float16* p_wv = panels + 4 * (size_t)PANEL_ELEMS;
  _Float16* p_wq = panels + 5 * (size_t)PANEL_ELEMS;
  _Float16* p_ow = panels + 6 * (size_t)PANEL_ELEMS;

  const int PB = (PANEL_ELEMS + 255) / 256;
  prep_bpanel<<<PB, 256, 0, stream>>>(w0, p_w0);
  prep_bpanel<<<PB, 256, 0, stream>>>(w1, p_w1);
  prep_bpanel<<<PB, 256, 0, stream>>>(w2, p_w2);
  prep_bpanel<<<PB, 256, 0, stream>>>(wk, p_wk);
  prep_bpanel<<<PB, 256, 0, stream>>>(wv, p_wv);
  prep_bpanel<<<PB, 256, 0, stream>>>(wq, p_wq);
  prep_bpanel<<<PB, 256, 0, stream>>>(ow, p_ow);

  hipMemsetAsync(deg, 0, (size_t)N * sizeof(float), stream);
  deg_kernel<<<(E + 255) / 256, 256, 0, stream>>>(dst, deg, E);
  dinv_kernel<<<(N + 255) / 256, 256, 0, stream>>>(deg, N);

  const _Float16* Pl[3] = {p_w0, p_w1, p_w2};
  const float*    Bl[3] = {b0, b1, b2};
  const float* cur_in = x;
  for (int l = 0; l < 3; ++l) {
    gemm128_wmma<<<(N + 31) / 32, 256, 0, stream>>>(cur_in, Pl[l], nullptr, s_hw, N);
    selfloop_init<<<((size_t)N * 32 + 255) / 256, 256, 0, stream>>>(s_hw, deg, s_agg, N * 32);
    edge_scatter<<<((size_t)E * 32 + 255) / 256, 256, 0, stream>>>(s_hw, src, dst, deg, s_agg, E);
    post_ln<<<((size_t)N * 32 + 255) / 256, 256, 0, stream>>>(s_agg, Bl[l], lng, lnb,
                                                              cur_in, s_cur, s_emb + (size_t)l * slab, N);
    cur_in = s_cur;
  }

  // k, v over all 3 layers (3N rows); q only for layer 2 (z[-1] is all that's used)
  gemm128_wmma<<<(3 * N + 31) / 32, 256, 0, stream>>>(s_emb, p_wk, nullptr, s_k, 3 * N);
  gemm128_wmma<<<(3 * N + 31) / 32, 256, 0, stream>>>(s_emb, p_wv, nullptr, s_v, 3 * N);
  gemm128_wmma<<<(N + 31) / 32, 256, 0, stream>>>(s_emb + 2 * slab, p_wq, nullptr, s_q, N);

  attn_ln<<<((size_t)N * 32 + 255) / 256, 256, 0, stream>>>(s_q, s_k, s_v, lng, lnb, s_z, N);

  gemm128_wmma<<<(N + 31) / 32, 256, 0, stream>>>(s_z, p_ow, ob, s_out, N);

  gather_rows<<<((size_t)4 * P * 32 + 255) / 256, 256, 0, stream>>>(s_out, dpos, tpos, dneg, tneg,
                                                                    (float*)d_out, P);
}
